// doublefusion_24352464569491
// MI455X (gfx1250) — compile-verified, matrix operand order checked by reading, not run on hardware
//
#include <hip/hip_runtime.h>
#include <hip/hip_bf16.h>
#include <math.h>

#define BDIM 32
#define CDIM 64
#define SDIM 112
#define SSZ  (SDIM*SDIM)          /* 12544 */
#define EPSV 1e-5f

typedef __attribute__((ext_vector_type(16))) __bf16  v16bf;
typedef __attribute__((ext_vector_type(8)))  __bf16  v8bf;
typedef __attribute__((ext_vector_type(4)))  __bf16  v4bf;
typedef __attribute__((ext_vector_type(8)))  float   v8f;
typedef __attribute__((ext_vector_type(4)))  float   f32x4;
typedef __attribute__((ext_vector_type(4)))  unsigned int u32x4;
typedef __attribute__((ext_vector_type(8)))  int     i32x8;

union Frag16 { v16bf f; v8bf h[2]; };

__device__ __forceinline__ float sigmoidf_(float x) { return 1.0f / (1.0f + __expf(-x)); }

__device__ __forceinline__ float blockReduceSum(float v, float* sbuf) {
  for (int off = 16; off > 0; off >>= 1) v += __shfl_xor(v, off, 32);
  int wave = threadIdx.x >> 5, lane = threadIdx.x & 31;
  if (lane == 0) sbuf[wave] = v;
  __syncthreads();
  int nw = blockDim.x >> 5;
  float r = (threadIdx.x < nw) ? sbuf[threadIdx.x] : 0.0f;
  if (wave == 0)
    for (int off = 8; off > 0; off >>= 1) r += __shfl_xor(r, off, 32);
  __syncthreads();
  return r;  // valid in thread 0
}

__device__ __forceinline__ float blockReduceMax(float v, float* sbuf) {
  for (int off = 16; off > 0; off >>= 1) v = fmaxf(v, __shfl_xor(v, off, 32));
  int wave = threadIdx.x >> 5, lane = threadIdx.x & 31;
  if (lane == 0) sbuf[wave] = v;
  __syncthreads();
  int nw = blockDim.x >> 5;
  float r = (threadIdx.x < nw) ? sbuf[threadIdx.x] : -1e30f;
  if (wave == 0)
    for (int off = 8; off > 0; off >>= 1) r = fmaxf(r, __shfl_xor(r, off, 32));
  __syncthreads();
  return r;
}

// ---------------------------------------------------------------------------
// Kernel 1: four 1x1 convs (channel GEMMs) via bf16 WMMA.
// Weights staged once per block as bf16 in LDS (A frags = 2x ds_load_b128);
// X staged transposed (pixel-major) per source in two phases
// (B frags = 2x ds_load_b128). grid (98, B), block 256 (8 waves).
// ---------------------------------------------------------------------------
__global__ __launch_bounds__(256) void k1_conv1x1(
    const float* __restrict__ rgb, const float* __restrict__ fre,
    const float* __restrict__ wq_fad, const float* __restrict__ bq_fad,
    const float* __restrict__ wq_lfs, const float* __restrict__ bq_lfs,
    const float* __restrict__ wk_fad, const float* __restrict__ bk_fad,
    const float* __restrict__ wk_lfs, const float* __restrict__ bk_lfs,
    __bf16* __restrict__ qF, __bf16* __restrict__ qL,
    __bf16* __restrict__ kF, __bf16* __restrict__ kL) {
  __shared__ __align__(16) __bf16 wsh[4][CDIM * CDIM];  // 32 KB
  __shared__ __align__(16) __bf16 xT[128 * CDIM];       // 16 KB [pix][ch]
  const int b = blockIdx.y;
  const int pix0 = blockIdx.x * 128;
  const int tid = threadIdx.x;

  const float* Ws[4] = {wq_fad, wk_fad, wq_lfs, wk_lfs};
  const float* Bs[4] = {bq_fad, bk_fad, bq_lfs, bk_lfs};
  __bf16* Os[4] = {qF, kF, qL, kL};

  // Stage all four weight matrices as bf16 (4096 float4 chunks).
  for (int j = tid; j < 4 * CDIM * CDIM / 4; j += 256) {
    const int t = j >> 10;  // 1024 float4 per 64x64 matrix
    const int off = (j & 1023) * 4;
    f32x4 w4 = *(const f32x4*)(Ws[t] + off);
    v4bf pk;
#pragma unroll
    for (int e = 0; e < 4; ++e) pk[e] = (__bf16)w4[e];
    *(v4bf*)(&wsh[t][off]) = pk;
  }

  const int wave = tid >> 5;
  const int lane = tid & 31;
  const int g = lane >> 4;
  const int n = lane & 15;
  const int pix = pix0 + wave * 16 + n;
  const int pxl = wave * 16 + n;

  for (int src = 0; src < 2; ++src) {
    __syncthreads();  // weights visible / xT free for overwrite
    const float* Xg = src ? fre : rgb;
    for (int i = tid; i < CDIM * 32; i += 256) {
      const int c = i >> 5;
      const int px = (i & 31) * 4;
      f32x4 x4 = *(const f32x4*)(Xg + ((size_t)b * CDIM + c) * SSZ + pix0 + px);
#pragma unroll
      for (int e = 0; e < 4; ++e) xT[(px + e) * CDIM + c] = (__bf16)x4[e];
    }
    __syncthreads();
    for (int tt = 0; tt < 2; ++tt) {
      const int t = src * 2 + tt;
      const float* bias = Bs[t];
      const __bf16* Wb = wsh[t];
      __bf16* O = Os[t];
#pragma unroll
      for (int mt = 0; mt < 4; ++mt) {
        v8f acc;
#pragma unroll
        for (int v = 0; v < 8; ++v) acc[v] = bias[mt * 16 + v + 8 * g];
        const int m = mt * 16 + n;
#pragma unroll
        for (int kt = 0; kt < 2; ++kt) {
          const int k1o = kt * 32 + 8 * g;
          const int k2o = k1o + 16;
          Frag16 a, bb;
          a.h[0] = *(const v8bf*)(Wb + m * CDIM + k1o);
          a.h[1] = *(const v8bf*)(Wb + m * CDIM + k2o);
          bb.h[0] = *(const v8bf*)(xT + pxl * CDIM + k1o);
          bb.h[1] = *(const v8bf*)(xT + pxl * CDIM + k2o);
          acc = __builtin_amdgcn_wmma_f32_16x16x32_bf16(false, a.f, false, bb.f,
                                                        (short)0, acc, false, false);
        }
#pragma unroll
        for (int v = 0; v < 8; ++v) {
          const int o = mt * 16 + v + 8 * g;
          O[((size_t)b * CDIM + o) * SSZ + pix] = (__bf16)acc[v];
        }
      }
    }
  }
}

// ---------------------------------------------------------------------------
// Kernel 2: attn = softmax(qF@kF^T + qL@kL^T) per (b,c) 112x112 matrix.
// kF/kL staged to LDS by the Tensor Data Mover (one wave issues two
// tensor_load_to_lds, S_WAIT_TENSORCNT, barrier). Fragments are 2x b128.
// grid 2048, block 224 (7 waves); wave w owns rows [16w,16w+16).
// ---------------------------------------------------------------------------
__global__ __launch_bounds__(224) void k2_attn(
    const __bf16* __restrict__ qF, const __bf16* __restrict__ qL,
    const __bf16* __restrict__ kF, const __bf16* __restrict__ kL,
    float* __restrict__ attn) {
  __shared__ __align__(16) __bf16 kf_s[SSZ];
  __shared__ __align__(16) __bf16 kl_s[SSZ];
  const int p = blockIdx.x;
  const size_t base = (size_t)p * SSZ;
  const int tid = threadIdx.x;

  if (tid < 32) {  // wave 0 issues two TDM copies: 1 row x 6272 dwords each
    const unsigned long long gaF = (unsigned long long)(const void*)(kF + base);
    const unsigned long long gaL = (unsigned long long)(const void*)(kL + base);
    const unsigned int ldsF = (unsigned int)(unsigned long long)(const void*)kf_s;
    const unsigned int ldsL = (unsigned int)(unsigned long long)(const void*)kl_s;
    // D# group1: data_size=4B, tensor_dim0=6272, tensor_dim1=1,
    // tile_dim0=6272, tile_dim1=1, dim0_stride=6272.
    i32x8 g1;
    g1[0] = 0x00020000;
    g1[1] = (int)((6272u & 0xFFFFu) << 16);
    g1[2] = (int)((6272u >> 16) | (1u << 16));
    g1[3] = (int)(6272u << 16);
    g1[4] = 1;
    g1[5] = 6272;
    g1[6] = 0;
    g1[7] = 0;
    u32x4 g0F = {1u, ldsF, (unsigned int)gaF,
                 (unsigned int)((gaF >> 32) & 0x01FFFFFFull) | 0x80000000u};
    u32x4 g0L = {1u, ldsL, (unsigned int)gaL,
                 (unsigned int)((gaL >> 32) & 0x01FFFFFFull) | 0x80000000u};
    asm volatile("tensor_load_to_lds %0, %1" ::"s"(g0F), "s"(g1) : "memory");
    asm volatile("tensor_load_to_lds %0, %1" ::"s"(g0L), "s"(g1) : "memory");
  }
  __builtin_amdgcn_s_wait_tensorcnt(0);
  __syncthreads();

  const int wave = tid >> 5;
  const int lane = tid & 31;
  const int g = lane >> 4;
  const int n = lane & 15;
  const __bf16* qFp = qF + base;
  const __bf16* qLp = qL + base;
  __builtin_prefetch(qFp + (size_t)(wave * 16) * SDIM, 0, 1);
  __builtin_prefetch(qLp + (size_t)(wave * 16) * SDIM, 0, 1);

  v8f acc[7] = {};
  const int i = wave * 16 + n;  // A-fragment row

  for (int kt = 0; kt < 7; ++kt) {
    const int k1o = kt * 32 + 8 * g;  // chunk starts are multiples of 8;
    const int k2o = k1o + 16;         // 112 % 16 == 0 -> no chunk straddles
    Frag16 a;
    a.h[0] = (k1o < SDIM) ? *(const v8bf*)(qFp + i * SDIM + k1o)
                          : *(const v8bf*)(qLp + i * SDIM + (k1o - SDIM));
    a.h[1] = (k2o < SDIM) ? *(const v8bf*)(qFp + i * SDIM + k2o)
                          : *(const v8bf*)(qLp + i * SDIM + (k2o - SDIM));
#pragma unroll
    for (int nt = 0; nt < 7; ++nt) {
      const int j = nt * 16 + n;  // B column; Mk[k][j] = kX[j][k] (row-major)
      Frag16 bb;
      bb.h[0] = (k1o < SDIM) ? *(const v8bf*)(kf_s + j * SDIM + k1o)
                             : *(const v8bf*)(kl_s + j * SDIM + (k1o - SDIM));
      bb.h[1] = (k2o < SDIM) ? *(const v8bf*)(kf_s + j * SDIM + k2o)
                             : *(const v8bf*)(kl_s + j * SDIM + (k2o - SDIM));
      acc[nt] = __builtin_amdgcn_wmma_f32_16x16x32_bf16(false, a.f, false, bb.f,
                                                        (short)0, acc[nt],
                                                        false, false);
    }
  }

  // Row softmax: row (wave*16 + v + 8g) lives in VGPR v of lane-half g.
#pragma unroll
  for (int v = 0; v < 8; ++v) {
    float mx = -1e30f;
#pragma unroll
    for (int nt = 0; nt < 7; ++nt) mx = fmaxf(mx, acc[nt][v]);
    for (int off = 1; off < 16; off <<= 1) mx = fmaxf(mx, __shfl_xor(mx, off, 16));
    float sum = 0.0f;
    float e[7];
#pragma unroll
    for (int nt = 0; nt < 7; ++nt) {
      e[nt] = __expf(acc[nt][v] - mx);
      sum += e[nt];
    }
    for (int off = 1; off < 16; off <<= 1) sum += __shfl_xor(sum, off, 16);
    const float inv = 1.0f / sum;
    const int row = wave * 16 + v + 8 * g;
#pragma unroll
    for (int nt = 0; nt < 7; ++nt)
      attn[base + (size_t)row * SDIM + nt * 16 + n] = e[nt] * inv;
  }
}

// ---------------------------------------------------------------------------
// Kernel 3: z1 = (fre*attn*s_lfs)*fad_cw+fad_cb ; z2 = (rgb*attn*s_fad)*lfs_cw+lfs_cb
// plus per-channel sum/sumsq (stats[0..255]).  grid (49, C, B), block 256.
// ---------------------------------------------------------------------------
__global__ __launch_bounds__(256) void k3_dw_reduce(
    const float* __restrict__ rgb, const float* __restrict__ fre,
    const float* __restrict__ attn,
    const float* __restrict__ fad_gamma, const float* __restrict__ lfs_gamma,
    const float* __restrict__ fad_cw, const float* __restrict__ fad_cb,
    const float* __restrict__ lfs_cw, const float* __restrict__ lfs_cb,
    float* __restrict__ z1, float* __restrict__ z2, float* __restrict__ stats) {
  __shared__ float sbuf[8];
  const int c = blockIdx.y, b = blockIdx.z;
  const size_t idx = ((size_t)b * CDIM + c) * SSZ + blockIdx.x * 256 + threadIdx.x;
  const float s_lfs = sigmoidf_(lfs_gamma[0]) * 2.0f - 1.0f;
  const float s_fad = sigmoidf_(fad_gamma[0]) * 2.0f - 1.0f;
  const float at = attn[idx];
  const float v1 = fre[idx] * at * s_lfs * fad_cw[c] + fad_cb[c];
  const float v2 = rgb[idx] * at * s_fad * lfs_cw[c] + lfs_cb[c];
  z1[idx] = v1;
  z2[idx] = v2;
  float s;
  s = blockReduceSum(v1, sbuf);        if (threadIdx.x == 0) atomicAdd(&stats[c], s);
  s = blockReduceSum(v1 * v1, sbuf);   if (threadIdx.x == 0) atomicAdd(&stats[64 + c], s);
  s = blockReduceSum(v2, sbuf);        if (threadIdx.x == 0) atomicAdd(&stats[128 + c], s);
  s = blockReduceSum(v2 * v2, sbuf);   if (threadIdx.x == 0) atomicAdd(&stats[192 + c], s);
}

// ---------------------------------------------------------------------------
// Kernel 4: y_fad = rgb + BN(z1), y_lfs = fre + BN(z2). float4 elementwise,
// flat grid over ELEMS/4 (planes are float4-aligned: SSZ % 4 == 0).
// ---------------------------------------------------------------------------
__global__ __launch_bounds__(256) void k4_bn_apply(
    const float* __restrict__ rgb, const float* __restrict__ fre,
    const float* __restrict__ z1, const float* __restrict__ z2,
    const float* __restrict__ stats,
    const float* __restrict__ fad_bg, const float* __restrict__ fad_bb,
    const float* __restrict__ lfs_bg, const float* __restrict__ lfs_bb,
    float* __restrict__ y_fad, float* __restrict__ y_lfs) {
  const size_t i4 = (size_t)blockIdx.x * 256 + threadIdx.x;
  const int c = (int)(i4 / (SSZ / 4)) & (CDIM - 1);
  const size_t idx = i4 * 4;
  const float N = (float)BDIM * (float)SSZ;
  const float m1 = stats[c] / N, va1 = stats[64 + c] / N - m1 * m1;
  const float m2 = stats[128 + c] / N, va2 = stats[192 + c] / N - m2 * m2;
  const float s1 = rsqrtf(va1 + EPSV) * fad_bg[c], b1 = fad_bb[c];
  const float s2 = rsqrtf(va2 + EPSV) * lfs_bg[c], b2 = lfs_bb[c];
  f32x4 r = *(const f32x4*)(rgb + idx);
  f32x4 f = *(const f32x4*)(fre + idx);
  f32x4 a1 = *(const f32x4*)(z1 + idx);
  f32x4 a2 = *(const f32x4*)(z2 + idx);
  f32x4 o1, o2;
#pragma unroll
  for (int e = 0; e < 4; ++e) {
    o1[e] = r[e] + (a1[e] - m1) * s1 + b1;
    o2[e] = f[e] + (a2[e] - m2) * s2 + b2;
  }
  *(f32x4*)(y_fad + idx) = o1;
  *(f32x4*)(y_lfs + idx) = o2;
}

// ---------------------------------------------------------------------------
// Kernel 5: pooled mean/max stats for k in {3,5,7}. grid (128, B), block 256.
// ---------------------------------------------------------------------------
__global__ __launch_bounds__(256) void k5_pool(
    const float* __restrict__ y_fad, const float* __restrict__ y_lfs,
    float* __restrict__ pavg, float* __restrict__ pmax) {
  __shared__ __align__(16) float plane[SSZ];
  __shared__ float sbuf[8];
  const int ch = blockIdx.x, b = blockIdx.y;
  const float* src = (ch < CDIM)
                         ? (y_fad + ((size_t)b * CDIM + ch) * SSZ)
                         : (y_lfs + ((size_t)b * CDIM + (ch - CDIM)) * SSZ);
  const f32x4* src4 = (const f32x4*)src;
  f32x4* pl4 = (f32x4*)plane;
  for (int i = threadIdx.x; i < SSZ / 4; i += 256) pl4[i] = src4[i];
  __syncthreads();
  const int kk[3] = {3, 5, 7};
  for (int kidx = 0; kidx < 3; ++kidx) {
    const int k = kk[kidx];
    const int nw = SDIM / k;
    const int nwin = nw * nw;
    const float invk2 = 1.0f / (float)(k * k);
    float ls = 0.0f, lm = -1e30f;
    for (int w = threadIdx.x; w < nwin; w += 256) {
      const int r0 = (w / nw) * k, c0 = (w % nw) * k;
      float s = 0.0f;
      for (int rr = 0; rr < k; ++rr)
        for (int cc = 0; cc < k; ++cc) s += plane[(r0 + rr) * SDIM + c0 + cc];
      const float avg = s * invk2;
      ls += avg;
      lm = fmaxf(lm, avg);
    }
    const float ts = blockReduceSum(ls, sbuf);
    const float tm = blockReduceMax(lm, sbuf);
    if (threadIdx.x == 0) {
      pavg[((size_t)b * 128 + ch) * 3 + kidx] = ts / (float)nwin;
      pmax[((size_t)b * 128 + ch) * 3 + kidx] = tm;
    }
    __syncthreads();
  }
}

// ---------------------------------------------------------------------------
// Kernel 6: channel-attention MLP. grid B, block 128.
// ---------------------------------------------------------------------------
__global__ __launch_bounds__(128) void k6_mlp(
    const float* __restrict__ pavg, const float* __restrict__ pmax,
    const float* __restrict__ lin1_w, const float* __restrict__ lin2_w,
    const float* __restrict__ mlp_w1, const float* __restrict__ mlp_w2,
    float* __restrict__ ca) {
  __shared__ float ap[128], mp[128], ha[16], hm[16];
  const int b = blockIdx.x, ch = threadIdx.x;
  float a = 0.0f, m = 0.0f;
  for (int kidx = 0; kidx < 3; ++kidx) {
    a += pavg[((size_t)b * 128 + ch) * 3 + kidx] * lin1_w[kidx];
    m += pmax[((size_t)b * 128 + ch) * 3 + kidx] * lin2_w[kidx];
  }
  ap[ch] = a;
  mp[ch] = m;
  __syncthreads();
  if (ch < 16) {
    float sa = 0.0f, sm = 0.0f;
    for (int c = 0; c < 128; ++c) {
      sa += mlp_w1[ch * 128 + c] * ap[c];
      sm += mlp_w1[ch * 128 + c] * mp[c];
    }
    ha[ch] = fmaxf(sa, 0.0f);
    hm[ch] = fmaxf(sm, 0.0f);
  }
  __syncthreads();
  float oa = 0.0f, om = 0.0f;
  for (int r = 0; r < 16; ++r) {
    oa += mlp_w2[ch * 16 + r] * ha[r];
    om += mlp_w2[ch * 16 + r] * hm[r];
  }
  ca[(size_t)b * 128 + ch] = sigmoidf_(oa + om);
}

// ---------------------------------------------------------------------------
// Kernel 7: z3 = (r_c*y_fad)*rgb_cw+rgb_cb ; z4 = (f_c*y_lfs)*fre_cw+fre_cb
// plus per-channel stats (stats[256..511]).
// ---------------------------------------------------------------------------
__global__ __launch_bounds__(256) void k7_dw_reduce2(
    const float* __restrict__ y_fad, const float* __restrict__ y_lfs,
    const float* __restrict__ ca,
    const float* __restrict__ rgb_cw, const float* __restrict__ rgb_cb,
    const float* __restrict__ fre_cw, const float* __restrict__ fre_cb,
    float* __restrict__ z3, float* __restrict__ z4, float* __restrict__ stats) {
  __shared__ float sbuf[8];
  const int c = blockIdx.y, b = blockIdx.z;
  const size_t idx = ((size_t)b * CDIM + c) * SSZ + blockIdx.x * 256 + threadIdx.x;
  const float rc = ca[(size_t)b * 128 + c];
  const float fc = ca[(size_t)b * 128 + CDIM + c];
  const float v3 = rc * y_fad[idx] * rgb_cw[c] + rgb_cb[c];
  const float v4 = fc * y_lfs[idx] * fre_cw[c] + fre_cb[c];
  z3[idx] = v3;
  z4[idx] = v4;
  float s;
  s = blockReduceSum(v3, sbuf);      if (threadIdx.x == 0) atomicAdd(&stats[256 + c], s);
  s = blockReduceSum(v3 * v3, sbuf); if (threadIdx.x == 0) atomicAdd(&stats[320 + c], s);
  s = blockReduceSum(v4, sbuf);      if (threadIdx.x == 0) atomicAdd(&stats[384 + c], s);
  s = blockReduceSum(v4 * v4, sbuf); if (threadIdx.x == 0) atomicAdd(&stats[448 + c], s);
}

// ---------------------------------------------------------------------------
// Kernel 8: rgb_out = y_fad + BN(z3), fre_out = y_lfs + BN(z4)  -> d_out
// float4 elementwise, flat grid over ELEMS/4.
// ---------------------------------------------------------------------------
__global__ __launch_bounds__(256) void k8_final(
    const float* __restrict__ y_fad, const float* __restrict__ y_lfs,
    const float* __restrict__ z3, const float* __restrict__ z4,
    const float* __restrict__ stats,
    const float* __restrict__ rgb_bg, const float* __restrict__ rgb_bb,
    const float* __restrict__ fre_bg, const float* __restrict__ fre_bb,
    float* __restrict__ out) {
  const size_t i4 = (size_t)blockIdx.x * 256 + threadIdx.x;
  const int c = (int)(i4 / (SSZ / 4)) & (CDIM - 1);
  const size_t idx = i4 * 4;
  const size_t half = (size_t)BDIM * CDIM * SSZ;
  const float N = (float)BDIM * (float)SSZ;
  const float m3 = stats[256 + c] / N, va3 = stats[320 + c] / N - m3 * m3;
  const float m4 = stats[384 + c] / N, va4 = stats[448 + c] / N - m4 * m4;
  const float s3 = rsqrtf(va3 + EPSV) * rgb_bg[c], b3 = rgb_bb[c];
  const float s4 = rsqrtf(va4 + EPSV) * fre_bg[c], b4 = fre_bb[c];
  f32x4 yf = *(const f32x4*)(y_fad + idx);
  f32x4 yl = *(const f32x4*)(y_lfs + idx);
  f32x4 a3 = *(const f32x4*)(z3 + idx);
  f32x4 a4 = *(const f32x4*)(z4 + idx);
  f32x4 o3, o4;
#pragma unroll
  for (int e = 0; e < 4; ++e) {
    o3[e] = yf[e] + (a3[e] - m3) * s3 + b3;
    o4[e] = yl[e] + (a4[e] - m4) * s4 + b4;
  }
  *(f32x4*)(out + idx) = o3;
  *(f32x4*)(out + half + idx) = o4;
}

// ---------------------------------------------------------------------------
// Host launcher
// ---------------------------------------------------------------------------
extern "C" void kernel_launch(void* const* d_in, const int* in_sizes, int n_in,
                              void* d_out, int out_size, void* d_ws, size_t ws_size,
                              hipStream_t stream) {
  (void)in_sizes; (void)n_in; (void)out_size; (void)ws_size;

  const float* rgb = (const float*)d_in[0];
  const float* fre = (const float*)d_in[1];
  const float* wq_fad = (const float*)d_in[2];
  const float* bq_fad = (const float*)d_in[3];
  const float* wq_lfs = (const float*)d_in[4];
  const float* bq_lfs = (const float*)d_in[5];
  const float* wk_fad = (const float*)d_in[6];
  const float* bk_fad = (const float*)d_in[7];
  const float* wk_lfs = (const float*)d_in[8];
  const float* bk_lfs = (const float*)d_in[9];
  const float* fad_gamma = (const float*)d_in[10];
  const float* lfs_gamma = (const float*)d_in[11];
  const float* fad_cw = (const float*)d_in[12];
  const float* fad_cb = (const float*)d_in[13];
  const float* fad_bg = (const float*)d_in[14];
  const float* fad_bb = (const float*)d_in[15];
  const float* lfs_cw = (const float*)d_in[16];
  const float* lfs_cb = (const float*)d_in[17];
  const float* lfs_bg = (const float*)d_in[18];
  const float* lfs_bb = (const float*)d_in[19];
  const float* lin1_w = (const float*)d_in[20];
  const float* lin2_w = (const float*)d_in[21];
  const float* mlp_w1 = (const float*)d_in[22];
  const float* mlp_w2 = (const float*)d_in[23];
  const float* rgb_cw = (const float*)d_in[24];
  const float* rgb_cb = (const float*)d_in[25];
  const float* rgb_bg = (const float*)d_in[26];
  const float* rgb_bb = (const float*)d_in[27];
  const float* fre_cw = (const float*)d_in[28];
  const float* fre_cb = (const float*)d_in[29];
  const float* fre_bg = (const float*)d_in[30];
  const float* fre_bb = (const float*)d_in[31];

  const size_t ELEMS = (size_t)BDIM * CDIM * SSZ;  // 25,690,112
  char* ws = (char*)d_ws;
  const size_t OFF_QF = 0;
  const size_t OFF_QL = OFF_QF + ELEMS * 2;
  const size_t OFF_KF = OFF_QL + ELEMS * 2;
  const size_t OFF_KL = OFF_KF + ELEMS * 2;
  const size_t OFF_ATTN = OFF_KL + ELEMS * 2;
  const size_t OFF_YFAD = OFF_ATTN + ELEMS * 4;
  const size_t OFF_YLFS = OFF_YFAD + ELEMS * 4;
  const size_t OFF_STAT = OFF_YLFS + ELEMS * 4;      // 512 floats
  const size_t OFF_PAVG = OFF_STAT + 512 * 4;        // B*128*3 floats
  const size_t OFF_PMAX = OFF_PAVG + (size_t)BDIM * 128 * 3 * 4;
  const size_t OFF_CA = OFF_PMAX + (size_t)BDIM * 128 * 3 * 4;
  // aliases (regions reused after their producers/consumers finish)
  const size_t OFF_Z1 = OFF_QF;               // over q/k (free after k2)
  const size_t OFF_Z2 = OFF_QF + ELEMS * 4;
  const size_t OFF_Z3 = OFF_ATTN;             // over attn (free after k3)
  const size_t OFF_Z4 = OFF_QF;               // over z1/z2 (free after k4)

  __bf16* qF = (__bf16*)(ws + OFF_QF);
  __bf16* qL = (__bf16*)(ws + OFF_QL);
  __bf16* kF = (__bf16*)(ws + OFF_KF);
  __bf16* kL = (__bf16*)(ws + OFF_KL);
  float* attn = (float*)(ws + OFF_ATTN);
  float* y_fad = (float*)(ws + OFF_YFAD);
  float* y_lfs = (float*)(ws + OFF_YLFS);
  float* stats = (float*)(ws + OFF_STAT);
  float* pavg = (float*)(ws + OFF_PAVG);
  float* pmax = (float*)(ws + OFF_PMAX);
  float* ca = (float*)(ws + OFF_CA);
  float* z1 = (float*)(ws + OFF_Z1);
  float* z2 = (float*)(ws + OFF_Z2);
  float* z3 = (float*)(ws + OFF_Z3);
  float* z4 = (float*)(ws + OFF_Z4);

  hipMemsetAsync(stats, 0, 512 * sizeof(float), stream);

  k1_conv1x1<<<dim3(SSZ / 128, BDIM), 256, 0, stream>>>(
      rgb, fre, wq_fad, bq_fad, wq_lfs, bq_lfs, wk_fad, bk_fad, wk_lfs, bk_lfs,
      qF, qL, kF, kL);

  k2_attn<<<dim3(BDIM * CDIM), 224, 0, stream>>>(qF, qL, kF, kL, attn);

  k3_dw_reduce<<<dim3(SSZ / 256, CDIM, BDIM), 256, 0, stream>>>(
      rgb, fre, attn, fad_gamma, lfs_gamma, fad_cw, fad_cb, lfs_cw, lfs_cb,
      z1, z2, stats);

  k4_bn_apply<<<dim3((unsigned)(ELEMS / 4 / 256)), 256, 0, stream>>>(
      rgb, fre, z1, z2, stats, fad_bg, fad_bb, lfs_bg, lfs_bb, y_fad, y_lfs);

  k5_pool<<<dim3(128, BDIM), 256, 0, stream>>>(y_fad, y_lfs, pavg, pmax);

  k6_mlp<<<dim3(BDIM), 128, 0, stream>>>(pavg, pmax, lin1_w, lin2_w, mlp_w1,
                                         mlp_w2, ca);

  k7_dw_reduce2<<<dim3(SSZ / 256, CDIM, BDIM), 256, 0, stream>>>(
      y_fad, y_lfs, ca, rgb_cw, rgb_cb, fre_cw, fre_cb, z3, z4, stats);

  k8_final<<<dim3((unsigned)(ELEMS / 4 / 256)), 256, 0, stream>>>(
      y_fad, y_lfs, z3, z4, stats, rgb_bg, rgb_bb, fre_bg, fre_bb,
      (float*)d_out);
}
